// Encoder_6_3710851744011
// MI455X (gfx1250) — matrix operand order repaired and checked by reading, not compiled, not run on hardware
//
#include <hip/hip_runtime.h>
#include <hip/hip_bf16.h>

typedef __attribute__((ext_vector_type(16))) __bf16 v16bf;
typedef __attribute__((ext_vector_type(8)))  float  v8f;
typedef __attribute__((ext_vector_type(4)))  unsigned v4u;
typedef __attribute__((ext_vector_type(8)))  int    v8i;
typedef __attribute__((ext_vector_type(4)))  int    v4i;

#define B_SZ 512
#define T_SZ 192
#define TP_SZ 200          // padded time rows: 2 halo + 192 + 6 tail
#define C_ENC 256
#define IP0 288            // layer0 input channels 257 padded to 288

// ---------------------------------------------------------------- TDM support
#if defined(__has_builtin)
#if __has_builtin(__builtin_amdgcn_tensor_load_to_lds) && \
    __has_builtin(__builtin_amdgcn_s_wait_tensorcnt)
#define HAVE_TDM 1
#endif
#endif
#ifndef HAVE_TDM
#define HAVE_TDM 0
#endif

#if HAVE_TDM
// 1-D contiguous global->LDS DMA of `nelem` bf16 elements via the Tensor Data
// Mover (CDNA5 ISA).  Issue from ONE wave only.
__device__ __forceinline__ void tdm_load_1d(void* ldsdst, const void* gsrc,
                                            unsigned nelem) {
    const unsigned lds_off = (unsigned)(uintptr_t)ldsdst;  // addr[31:0] = LDS offset
    const unsigned long long ga = (unsigned long long)(uintptr_t)gsrc;
    // group0: count=1 | lds_addr | global_addr[56:0] | type=2 (bits 127:126)
    v4u g0 = { 1u,
               lds_off,
               (unsigned)(ga & 0xffffffffu),
               (unsigned)((ga >> 32) & 0x01ffffffu) | 0x80000000u };
    // group1: wg_mask=0, data_size=2B; tensor_dim0 = tile_dim0 = nelem (1-D),
    // tensor_dim1 = 1, tensor_dim0_stride = nelem.
    v8i g1 = { (int)0x00010000,
               (int)((nelem & 0xffffu) << 16),          // tensor_dim0[15:0] @ bit48
               (int)((nelem >> 16) | (1u << 16)),       // dim0[31:16] | tensor_dim1=1
               (int)((nelem & 0xffffu) << 16),          // tile_dim0 @ bit112
               0,                                       // tile_dim1=0, tile_dim2=0
               (int)nelem,                              // tensor_dim0_stride lo
               0, 0 };
    v4i z = { 0, 0, 0, 0 };
#if __clang_major__ >= 23
    v8i z8 = { 0, 0, 0, 0, 0, 0, 0, 0 };
    __builtin_amdgcn_tensor_load_to_lds(g0, g1, z, z, z8, 0);
#else
    __builtin_amdgcn_tensor_load_to_lds(g0, g1, z, z, 0);
#endif
}
#endif

// ---------------------------------------------------------------- threefry2x32
__device__ __forceinline__ unsigned rotl32(unsigned x, int r) {
    return (x << r) | (x >> (32 - r));
}
__device__ void threefry2x32(unsigned k0, unsigned k1, unsigned x0, unsigned x1,
                             unsigned& o0, unsigned& o1) {
    unsigned ks0 = k0, ks1 = k1, ks2 = k0 ^ k1 ^ 0x1BD11BDAu;
    unsigned X0 = x0 + ks0, X1 = x1 + ks1;
    const int R0[4] = {13, 15, 26, 6};
    const int R1[4] = {17, 29, 16, 24};
#pragma unroll
    for (int blk = 0; blk < 5; ++blk) {
#pragma unroll
        for (int r = 0; r < 4; ++r) {
            int rot = (blk & 1) ? R1[r] : R0[r];
            X0 += X1; X1 = rotl32(X1, rot); X1 ^= X0;
        }
        unsigned i = blk + 1;
        unsigned a = (i % 3 == 0) ? ks0 : (i % 3 == 1) ? ks1 : ks2;
        unsigned b = ((i + 1) % 3 == 0) ? ks0 : ((i + 1) % 3 == 1) ? ks1 : ks2;
        X0 += a; X1 += b + i;
    }
    o0 = X0; o1 = X1;
}

// ---------------------------------------------------------------- weight packing
// wpk layout: [k(5)][mt(16)][it(Itiles)][lane(32)][e(16)] bf16, A-fragment order.
__global__ __launch_bounds__(256) void pack_conv_w(const float* __restrict__ w,
                                                   __bf16* __restrict__ wpk,
                                                   int I, int Itiles, int total) {
    int idx = blockIdx.x * 256 + threadIdx.x;
    if (idx >= total) return;
    int e    = idx & 15;
    int lane = (idx >> 4) & 31;
    int rest = idx >> 9;
    int it   = rest % Itiles; rest /= Itiles;
    int mt   = rest & 15;
    int k    = rest >> 4;
    int row  = lane & 15, hi = lane >> 4;
    int o    = mt * 16 + row;
    int e2   = hi * 16 + e;
    int ki   = it * 32 + e2;
    float v  = (ki < I) ? w[((size_t)o * I + ki) * 5 + k] : 0.f;
    wpk[idx] = (__bf16)v;
}

// wpkL layout: [dir(2)][mt(8)][it(8)][lane(32)][e(16)]
__global__ __launch_bounds__(256) void pack_lstm_w(const float* __restrict__ wf,
                                                   const float* __restrict__ wb,
                                                   __bf16* __restrict__ wpk) {
    int idx = blockIdx.x * 256 + threadIdx.x;
    if (idx >= 2 * 8 * 8 * 512) return;
    int e    = idx & 15;
    int lane = (idx >> 4) & 31;
    int it   = (idx >> 9) & 7;
    int mt   = (idx >> 12) & 7;
    int dir  = (idx >> 15) & 1;
    const float* w = dir ? wb : wf;
    int row = lane & 15, hi = lane >> 4;
    int g   = mt * 16 + row;
    int e2  = hi * 16 + e;
    int ki  = it * 32 + e2;
    wpk[idx] = (__bf16)w[(size_t)g * C_ENC + ki];
}

// ------------------------------------------------------- x -> bf16 time-major, padded
__global__ __launch_bounds__(256) void x_to_bf16(const float* __restrict__ x,
                                                 __bf16* __restrict__ xb0) {
    size_t idx = (size_t)blockIdx.x * 256 + threadIdx.x;
    const size_t total = (size_t)B_SZ * TP_SZ * IP0;
    if (idx >= total) return;
    int ip = idx % IP0;
    size_t r = idx / IP0;
    int tp = r % TP_SZ;
    int b  = r / TP_SZ;
    float v = 0.f;
    if (ip < 257 && tp >= 2 && tp < 194)
        v = x[((size_t)b * 257 + ip) * T_SZ + (tp - 2)];
    xb0[idx] = (__bf16)v;
}

// ---------------------------------------------------------------- conv1d via WMMA
// out[o,t] = sum_k sum_i W[o,i,k] * x[i, t+k-2]   as 5 shifted GEMMs.
// Each wave owns two M-tiles (mt, mt+8) sharing every B fragment: two
// independent accumulator chains hide WMMA latency and halve LDS traffic.
template <int IPAD, int ITILES>
__global__ __launch_bounds__(256) void conv_wmma(const __bf16* __restrict__ xb,
                                                 const __bf16* __restrict__ wpk,
                                                 const float* __restrict__ bias,
                                                 __bf16* __restrict__ tmp) {
    extern __shared__ char smem_raw[];
    __bf16* lds = reinterpret_cast<__bf16*>(smem_raw);   // [68][IPAD] time-major
    __shared__ float sbias[C_ENC];
    const int b = blockIdx.x;
    const int tbase = blockIdx.y * 64;                   // 3 chunks of 64 t
    const __bf16* src = xb + ((size_t)b * TP_SZ + tbase) * IPAD;

#if HAVE_TDM
    if (threadIdx.x < 32) {                              // wave 0 issues the DMA
        tdm_load_1d(lds, src, 68 * IPAD);
    }
#else
    {
        const uint4* s4 = reinterpret_cast<const uint4*>(src);
        uint4* d4 = reinterpret_cast<uint4*>(lds);
        for (int i = threadIdx.x; i < 68 * IPAD * 2 / 16; i += 256) d4[i] = s4[i];
    }
#endif
    sbias[threadIdx.x] = bias[threadIdx.x];              // overlaps with DMA
#if HAVE_TDM
    if (threadIdx.x < 32) __builtin_amdgcn_s_wait_tensorcnt(0);
#endif
    __syncthreads();

    const int lane  = threadIdx.x & 31;
    const int wave  = threadIdx.x >> 5;
    const int nlane = lane & 15;
    const int khalf = lane >> 4;
    const int mt0 = wave;                                // 0..7
    const int mt1 = wave + 8;                            // 8..15

    const __bf16* a0base = wpk + ((size_t)mt0 * ITILES * 32 + lane) * 16;
    const __bf16* a1base = wpk + ((size_t)mt1 * ITILES * 32 + lane) * 16;

#pragma unroll 1
    for (int nt = 0; nt < 4; ++nt) {
        v8f acc0 = {}, acc1 = {};
        const __bf16* bbase = lds + (size_t)(nt * 16 + nlane) * IPAD + khalf * 16;
#pragma unroll
        for (int k = 0; k < 5; ++k) {
#pragma unroll
            for (int it = 0; it < ITILES; ++it) {
                const size_t aoff = (size_t)k * 16 * ITILES * 512 + (size_t)it * 512;
                v16bf bb = *reinterpret_cast<const v16bf*>(
                    bbase + (size_t)k * IPAD + it * 32);
                v16bf a0 = *reinterpret_cast<const v16bf*>(a0base + aoff);
                v16bf a1 = *reinterpret_cast<const v16bf*>(a1base + aoff);
                acc0 = __builtin_amdgcn_wmma_f32_16x16x32_bf16(
                    false, a0, false, bb, (short)0, acc0, false, false);
                acc1 = __builtin_amdgcn_wmma_f32_16x16x32_bf16(
                    false, a1, false, bb, (short)0, acc1, false, false);
            }
        }
        const int t = tbase + nt * 16 + nlane;
#pragma unroll
        for (int v = 0; v < 8; ++v) {
            const int o0 = mt0 * 16 + khalf * 8 + v;
            const int o1 = mt1 * 16 + khalf * 8 + v;
            tmp[((size_t)b * C_ENC + o0) * T_SZ + t] = (__bf16)(acc0[v] + sbias[o0]);
            tmp[((size_t)b * C_ENC + o1) * T_SZ + t] = (__bf16)(acc1[v] + sbias[o1]);
        }
    }
}

// ---------------------------------------------------------------- GroupNorm stats
__global__ __launch_bounds__(128) void gn_stats(const __bf16* __restrict__ tmp,
                                                float* __restrict__ stats) {
    const int b = blockIdx.x, g = blockIdx.y;
    const __bf16* p = tmp + ((size_t)b * C_ENC + g * 16) * T_SZ;
    float s = 0.f, ss = 0.f;
    for (int i = threadIdx.x; i < 16 * T_SZ; i += 128) {
        float v = (float)p[i];
        s += v; ss += v * v;
    }
    __shared__ float rs[128], rq[128];
    rs[threadIdx.x] = s; rq[threadIdx.x] = ss;
    __syncthreads();
    for (int o = 64; o > 0; o >>= 1) {
        if (threadIdx.x < o) {
            rs[threadIdx.x] += rs[threadIdx.x + o];
            rq[threadIdx.x] += rq[threadIdx.x + o];
        }
        __syncthreads();
    }
    if (threadIdx.x == 0) {
        const float inv = 1.f / (16.f * T_SZ);
        float mu = rs[0] * inv;
        float var = rq[0] * inv - mu * mu;
        stats[(b * 16 + g) * 2 + 0] = mu;
        stats[(b * 16 + g) * 2 + 1] = rsqrtf(var + 1e-5f);
    }
}

// --------------------------------------------- GN apply + ReLU + transpose to [t][c]
__global__ __launch_bounds__(256) void gn_apply(const __bf16* __restrict__ tmp,
                                                const float* __restrict__ stats,
                                                const float* __restrict__ gw,
                                                const float* __restrict__ gb,
                                                __bf16* __restrict__ act) {
    const int b = blockIdx.x;
    const int tbase = blockIdx.y * 32;                   // 6 tiles of 32 t
    __shared__ float mu_s[16], rs_s[16], gw_s[256], gb_s[256];
    __shared__ __bf16 tile[32][260];
    if (threadIdx.x < 16) {
        mu_s[threadIdx.x] = stats[(b * 16 + threadIdx.x) * 2 + 0];
        rs_s[threadIdx.x] = stats[(b * 16 + threadIdx.x) * 2 + 1];
    }
    gw_s[threadIdx.x] = gw[threadIdx.x];
    gb_s[threadIdx.x] = gb[threadIdx.x];
    __syncthreads();
    const int tl = threadIdx.x & 31;
    const int cw = threadIdx.x >> 5;
#pragma unroll 4
    for (int i = 0; i < 32; ++i) {
        const int c = i * 8 + cw, g = c >> 4;
        float v = (float)tmp[((size_t)b * C_ENC + c) * T_SZ + tbase + tl];
        v = (v - mu_s[g]) * rs_s[g] * gw_s[c] + gb_s[c];
        tile[tl][c] = (__bf16)fmaxf(v, 0.f);
    }
    __syncthreads();
#pragma unroll 4
    for (int i = 0; i < 32; ++i)
        act[((size_t)b * T_SZ + tbase + i) * C_ENC + threadIdx.x] = tile[i][threadIdx.x];
}

// ---------------------------------------------------------------- interp_lnr
// act: [B][192][256] bf16  ->  xbn: [B][200][256] bf16 (halo rows zero)
__global__ __launch_bounds__(256) void interp_kernel(const __bf16* __restrict__ act,
                                                     __bf16* __restrict__ xbn,
                                                     int layer) {
    const int b = blockIdx.x;
    __shared__ int   gi_s[448];
    __shared__ float lam_s[448];
    __shared__ int   pos_s[448];
    if (threadIdx.x == 0) {
        unsigned lk0, lk1;
        threefry2x32(0u, 42u, 0u, (unsigned)layer, lk0, lk1);
        int offset = 0, pos = 0;
        for (int s = 0; s < 7; ++s) {
            unsigned r0, r1;
            threefry2x32(lk0, lk1, (unsigned)(b * 7 + s), 0x9E3779B9u, r0, r1);
            const float scale = 0.5f + (float)(r0 >> 8) * (1.f / 16777216.f);
            const int   lseg  = 19 + (int)(r1 % 13u);
            for (int l = 0; l < 64; ++l) {
                const int jj = s * 64 + l;
                float idxf = (float)l / scale;
                float fl = floorf(idxf);
                float lam = idxf - fl;
                int iorg = (int)fl + offset;
                bool keep = (fl < (float)(lseg - 1)) && (iorg < T_SZ - 1);
                if (keep && pos < T_SZ) {
                    int g = iorg < 0 ? 0 : (iorg > T_SZ - 2 ? T_SZ - 2 : iorg);
                    gi_s[jj] = g; lam_s[jj] = lam; pos_s[jj] = pos;
                } else {
                    pos_s[jj] = -1;
                }
                if (keep) ++pos;
            }
            offset += lseg;
        }
    }
    {
        unsigned* z = reinterpret_cast<unsigned*>(xbn + (size_t)b * TP_SZ * C_ENC);
        for (int i = threadIdx.x; i < TP_SZ * C_ENC / 2; i += 256) z[i] = 0u;
    }
    __syncthreads();
    const int c = threadIdx.x;
    const __bf16* ab = act + (size_t)b * T_SZ * C_ENC;
    __bf16* ob = xbn + (size_t)b * TP_SZ * C_ENC;
#pragma unroll 1
    for (int jj = 0; jj < 448; ++jj) {
        const int p = pos_s[jj];
        if (p >= 0) {
            const int g = gi_s[jj];
            const float lam = lam_s[jj];
            float v = (1.f - lam) * (float)ab[(size_t)g * C_ENC + c] +
                      lam * (float)ab[(size_t)(g + 1) * C_ENC + c];
            ob[(size_t)(2 + p) * C_ENC + c] = (__bf16)v;
        }
    }
}

// ---------------------------------------------------------- LSTM gate GEMM (WMMA)
// xg: [dir(2)][B][192][128] bf16 = x @ w_ih^T + b_ih + b_hh
// Each wave owns (dir0, mt=wave) and (dir1, mt=wave) sharing B fragments.
__global__ __launch_bounds__(256) void lstm_gates(const __bf16* __restrict__ xb,
                                                  const __bf16* __restrict__ wpk,
                                                  const float* __restrict__ bih_f,
                                                  const float* __restrict__ bhh_f,
                                                  const float* __restrict__ bih_b,
                                                  const float* __restrict__ bhh_b,
                                                  __bf16* __restrict__ xg) {
    __shared__ __bf16 lds[64 * C_ENC];                   // 32 KB
    const int b = blockIdx.x;
    const int tbase = blockIdx.y * 64;
    const __bf16* src = xb + ((size_t)b * TP_SZ + 2 + tbase) * C_ENC;
#if HAVE_TDM
    if (threadIdx.x < 32) {
        tdm_load_1d(lds, src, 64 * C_ENC);
        __builtin_amdgcn_s_wait_tensorcnt(0);
    }
#else
    {
        const uint4* s4 = reinterpret_cast<const uint4*>(src);
        uint4* d4 = reinterpret_cast<uint4*>(lds);
        for (int i = threadIdx.x; i < 64 * C_ENC * 2 / 16; i += 256) d4[i] = s4[i];
    }
#endif
    __syncthreads();
    const int lane  = threadIdx.x & 31;
    const int wave  = threadIdx.x >> 5;
    const int nlane = lane & 15;
    const int khalf = lane >> 4;
    const int mt    = wave;                              // 0..7

    const __bf16* a0base = wpk + ((size_t)mt * 8 * 32 + lane) * 16;            // dir0
    const __bf16* a1base = wpk + (((size_t)(8 + mt) * 8) * 32 + lane) * 16;    // dir1

#pragma unroll 1
    for (int nt = 0; nt < 4; ++nt) {
        v8f acc0 = {}, acc1 = {};
        const __bf16* brow = lds + (size_t)(nt * 16 + nlane) * C_ENC + khalf * 16;
#pragma unroll
        for (int it = 0; it < 8; ++it) {
            v16bf bb = *reinterpret_cast<const v16bf*>(brow + it * 32);
            v16bf a0 = *reinterpret_cast<const v16bf*>(a0base + (size_t)it * 512);
            v16bf a1 = *reinterpret_cast<const v16bf*>(a1base + (size_t)it * 512);
            acc0 = __builtin_amdgcn_wmma_f32_16x16x32_bf16(
                false, a0, false, bb, (short)0, acc0, false, false);
            acc1 = __builtin_amdgcn_wmma_f32_16x16x32_bf16(
                false, a1, false, bb, (short)0, acc1, false, false);
        }
        const int t = tbase + nt * 16 + nlane;
#pragma unroll
        for (int v = 0; v < 8; ++v) {
            const int g = mt * 16 + khalf * 8 + v;
            xg[(((size_t)0 * B_SZ + b) * T_SZ + t) * 128 + g] =
                (__bf16)(acc0[v] + bih_f[g] + bhh_f[g]);
            xg[(((size_t)1 * B_SZ + b) * T_SZ + t) * 128 + g] =
                (__bf16)(acc1[v] + bih_b[g] + bhh_b[g]);
        }
    }
}

// ---------------------------------------------------------------- LSTM recurrence
__device__ __forceinline__ float sigmoidf(float x) { return 1.f / (1.f + __expf(-x)); }

__global__ __launch_bounds__(32) void lstm_rec(const __bf16* __restrict__ xg,
                                               const float* __restrict__ whh_f,
                                               const float* __restrict__ whh_b,
                                               float* __restrict__ out) {
    const int b = blockIdx.x;
    const int dir = blockIdx.y;
    const int j = threadIdx.x;                           // hidden unit 0..31
    __shared__ float wT[32][128];                        // wT[k][g]
    __shared__ float hsh[32];
    const float* whh = dir ? whh_b : whh_f;              // [128][32]
    for (int idx = j; idx < 4096; idx += 32)
        wT[idx & 31][idx >> 5] = whh[idx];
    hsh[j] = 0.f;
    __syncthreads();
    float h = 0.f, c = 0.f;
    for (int step = 0; step < T_SZ; ++step) {
        const int t = dir ? (T_SZ - 1 - step) : step;
        const __bf16* gp = xg + (((size_t)dir * B_SZ + b) * T_SZ + t) * 128;
        float gi = (float)gp[j], gf = (float)gp[32 + j];
        float gg = (float)gp[64 + j], go = (float)gp[96 + j];
#pragma unroll 8
        for (int k = 0; k < 32; ++k) {
            const float hk = hsh[k];
            gi += wT[k][j] * hk;
            gf += wT[k][32 + j] * hk;
            gg += wT[k][64 + j] * hk;
            go += wT[k][96 + j] * hk;
        }
        c = sigmoidf(gf) * c + sigmoidf(gi) * tanhf(gg);
        h = sigmoidf(go) * tanhf(c);
        __syncthreads();
        hsh[j] = h;
        __syncthreads();
        if (!dir && ((t & 7) == 7))
            out[((size_t)b * 24 + (t >> 3)) * 64 + j] = h;
        if (dir && ((t & 7) == 0))
            out[((size_t)b * 24 + (t >> 3)) * 64 + 32 + j] = h;
    }
}

// ---------------------------------------------------------------- host launcher
extern "C" void kernel_launch(void* const* d_in, const int* in_sizes, int n_in,
                              void* d_out, int out_size, void* d_ws, size_t ws_size,
                              hipStream_t stream) {
    const float* x    = (const float*)d_in[0];
    const float* cw0  = (const float*)d_in[1];
    const float* cb0  = (const float*)d_in[2];
    const float* gw0  = (const float*)d_in[3];
    const float* gb0  = (const float*)d_in[4];
    const float* cw1  = (const float*)d_in[5];
    const float* cb1  = (const float*)d_in[6];
    const float* gw1  = (const float*)d_in[7];
    const float* gb1  = (const float*)d_in[8];
    const float* cw2  = (const float*)d_in[9];
    const float* cb2  = (const float*)d_in[10];
    const float* gw2  = (const float*)d_in[11];
    const float* gb2  = (const float*)d_in[12];
    const float* wihf = (const float*)d_in[13];
    const float* whhf = (const float*)d_in[14];
    const float* bihf = (const float*)d_in[15];
    const float* bhhf = (const float*)d_in[16];
    const float* wihb = (const float*)d_in[17];
    const float* whhb = (const float*)d_in[18];
    const float* bihb = (const float*)d_in[19];
    const float* bhhb = (const float*)d_in[20];
    float* out = (float*)d_out;

    char* ws = (char*)d_ws;
    size_t off = 0;
    auto take = [&](size_t bytes) {
        char* p = ws + off;
        off = (off + bytes + 255) & ~(size_t)255;
        return p;
    };
    __bf16* wpk0 = (__bf16*)take((size_t)5 * 16 * 9 * 512 * 2);
    __bf16* wpk1 = (__bf16*)take((size_t)5 * 16 * 8 * 512 * 2);
    __bf16* wpk2 = (__bf16*)take((size_t)5 * 16 * 8 * 512 * 2);
    __bf16* wpkL = (__bf16*)take((size_t)2 * 8 * 8 * 512 * 2);
    __bf16* xb0  = (__bf16*)take((size_t)B_SZ * TP_SZ * IP0 * 2);
    __bf16* xbA  = (__bf16*)take((size_t)B_SZ * TP_SZ * C_ENC * 2);
    __bf16* xbB  = (__bf16*)take((size_t)B_SZ * TP_SZ * C_ENC * 2);
    __bf16* tmp  = (__bf16*)take((size_t)B_SZ * C_ENC * T_SZ * 2); // aliased as xg
    __bf16* act  = (__bf16*)take((size_t)B_SZ * T_SZ * C_ENC * 2);
    float*  stats = (float*)take((size_t)B_SZ * 16 * 2 * 4);

    // ---- packing + input conversion
    {
        int tot0 = 5 * 16 * 9 * 512;
        pack_conv_w<<<(tot0 + 255) / 256, 256, 0, stream>>>(cw0, wpk0, 257, 9, tot0);
        int tot1 = 5 * 16 * 8 * 512;
        pack_conv_w<<<(tot1 + 255) / 256, 256, 0, stream>>>(cw1, wpk1, 256, 8, tot1);
        pack_conv_w<<<(tot1 + 255) / 256, 256, 0, stream>>>(cw2, wpk2, 256, 8, tot1);
        pack_lstm_w<<<(2 * 8 * 8 * 512 + 255) / 256, 256, 0, stream>>>(wihf, wihb, wpkL);
        size_t totx = (size_t)B_SZ * TP_SZ * IP0;
        x_to_bf16<<<(unsigned)((totx + 255) / 256), 256, 0, stream>>>(x, xb0);
    }

    // ---- layer 0
    conv_wmma<IP0, 9><<<dim3(B_SZ, 3), 256, 68 * IP0 * 2, stream>>>(xb0, wpk0, cb0, tmp);
    gn_stats<<<dim3(B_SZ, 16), 128, 0, stream>>>(tmp, stats);
    gn_apply<<<dim3(B_SZ, 6), 256, 0, stream>>>(tmp, stats, gw0, gb0, act);
    interp_kernel<<<B_SZ, 256, 0, stream>>>(act, xbA, 0);

    // ---- layer 1
    conv_wmma<C_ENC, 8><<<dim3(B_SZ, 3), 256, 68 * C_ENC * 2, stream>>>(xbA, wpk1, cb1, tmp);
    gn_stats<<<dim3(B_SZ, 16), 128, 0, stream>>>(tmp, stats);
    gn_apply<<<dim3(B_SZ, 6), 256, 0, stream>>>(tmp, stats, gw1, gb1, act);
    interp_kernel<<<B_SZ, 256, 0, stream>>>(act, xbB, 1);

    // ---- layer 2
    conv_wmma<C_ENC, 8><<<dim3(B_SZ, 3), 256, 68 * C_ENC * 2, stream>>>(xbB, wpk2, cb2, tmp);
    gn_stats<<<dim3(B_SZ, 16), 128, 0, stream>>>(tmp, stats);
    gn_apply<<<dim3(B_SZ, 6), 256, 0, stream>>>(tmp, stats, gw2, gb2, act);
    interp_kernel<<<B_SZ, 256, 0, stream>>>(act, xbA, 2);

    // ---- LSTM (xg aliases tmp: identical byte size)
    __bf16* xg = tmp;
    lstm_gates<<<dim3(B_SZ, 3), 256, 0, stream>>>(xbA, wpkL, bihf, bhhf, bihb, bhhb, xg);
    lstm_rec<<<dim3(B_SZ, 2), 32, 0, stream>>>(xg, whhf, whhb, out);
}
